// GraphRecurrent_58179626991923
// MI455X (gfx1250) — matrix operand order repaired and checked by reading, not compiled, not run on hardware
//
#include <hip/hip_runtime.h>
#include <hip/hip_bf16.h>

#define Nn 50000
#define Ee 600000
#define Hh 128

typedef __attribute__((ext_vector_type(16))) _Float16 v16h;
typedef __attribute__((ext_vector_type(8)))  _Float16 v8h;
typedef __attribute__((ext_vector_type(8)))  float    v8f;

__device__ __forceinline__ float sigmoidf_(float x) { return 1.0f / (1.0f + __expf(-x)); }

// ---------------------------------------------------------------------------
// Zero the accumulation regions of the workspace (aggr1, p, q, aggr3).
__global__ __launch_bounds__(256) void zero_ws_kernel(float* __restrict__ ws, long n) {
  long i = (long)blockIdx.x * blockDim.x + threadIdx.x;
  long stride = (long)gridDim.x * blockDim.x;
  for (; i < n; i += stride) ws[i] = 0.0f;
}

// ---------------------------------------------------------------------------
// Precompute the rank-1-collapsed constants:
//   u[g] = W_ih[g,:] . nn1_w     v[g] = W_ih[g,:] . nn1_b + b_ih[g] + b_hh[g]
//   scal = {alpha, beta} for e1[e] = alpha*edge_attr[e] + beta
//   W16  = f16 copy of conv3_nn_w ([j][k] row-major == WMMA B column-major)
__global__ __launch_bounds__(512) void prep_kernel(
    const float* __restrict__ emw, const float* __restrict__ emb,
    const float* __restrict__ lin_w, const float* __restrict__ lin_b,
    const float* __restrict__ nn1_w, const float* __restrict__ nn1_b,
    const float* __restrict__ w_ih, const float* __restrict__ b_ih,
    const float* __restrict__ b_hh, const float* __restrict__ w3,
    float* __restrict__ u, float* __restrict__ v,
    float* __restrict__ scal, _Float16* __restrict__ W16)
{
  int t = threadIdx.x;
  if (t < 4 * Hh) {
    float su = 0.f, sv = 0.f;
    for (int k = 0; k < Hh; ++k) {
      float w = w_ih[t * Hh + k];
      su += w * nn1_w[k];
      sv += w * nn1_b[k];
    }
    u[t] = su;
    v[t] = sv + b_ih[t] + b_hh[t];
  }
  for (int i = t; i < Hh * Hh; i += 512) W16[i] = (_Float16)w3[i];
  if (t == 0) {
    float a = 0.f, b = 0.f;
    for (int h = 0; h < Hh; ++h) { a += emw[h] * lin_w[h]; b += emb[h] * lin_w[h]; }
    scal[0] = a;
    scal[1] = b + lin_b[0];
  }
}

// ---------------------------------------------------------------------------
// GINEConv1 message pass: msg = relu(x[src] + alpha*a[e] + beta), scatter-add.
__global__ __launch_bounds__(256) void edge1_kernel(
    const float* __restrict__ x, const float* __restrict__ ea,
    const int* __restrict__ ei, const float* __restrict__ scal,
    float* __restrict__ aggr1)
{
  int e = blockIdx.x * 256 + threadIdx.x;
  if (e >= Ee) return;
  int src = ei[e], dst = ei[Ee + e];
  float m = fmaxf(x[src] + scal[0] * ea[e] + scal[1], 0.f);
  atomicAdd(&aggr1[dst], m);
}

// ---------------------------------------------------------------------------
// conv1-nn + LSTM step, fully elementwise thanks to the rank-1 collapse.
// gates[n,g] = s[n]*u[g] + v[g];  c = sig(i)*tanh(g);  hn = sig(o)*tanh(c)
__global__ __launch_bounds__(256) void node_lstm_kernel(
    const float* __restrict__ x, const float* __restrict__ aggr1,
    const float* __restrict__ u, const float* __restrict__ v,
    float* __restrict__ hn_out, float* __restrict__ c_out)
{
  int t = blockIdx.x * 256 + threadIdx.x;  // t in [0, N*H)
  int n = t >> 7, k = t & (Hh - 1);
  float s  = x[n] + aggr1[n];
  float gi = s * u[k]            + v[k];             // input gate
  float gg = s * u[2 * Hh + k]   + v[2 * Hh + k];    // cell gate
  float go = s * u[3 * Hh + k]   + v[3 * Hh + k];    // output gate
  float c  = sigmoidf_(gi) * tanhf(gg);              // f*c0 == 0
  float hn = sigmoidf_(go) * tanhf(c);
  hn_out[t] = hn;
  c_out[t]  = c;
}

// ---------------------------------------------------------------------------
// GINEConv3 message pass (the bandwidth roof): one wave per edge, 4 cols/lane.
// msg3 = relu(hn[src,k] + a[e]*emw[k] + emb[k]); scatter-add into aggr3[dst].
__global__ __launch_bounds__(256) void edge2_kernel(
    const float* __restrict__ hn, const float* __restrict__ ea,
    const int* __restrict__ ei, const float* __restrict__ emw,
    const float* __restrict__ emb, float* __restrict__ aggr3)
{
  int e = blockIdx.x * 8 + (threadIdx.x >> 5);
  if (e >= Ee) return;
  int lane = threadIdx.x & 31;
  int k0 = lane * 4;
  int src = ei[e], dst = ei[Ee + e];
  float a = ea[e];
  float4 hv = *(const float4*)(hn + (long)src * Hh + k0);
  float4 w  = *(const float4*)(emw + k0);
  float4 b  = *(const float4*)(emb + k0);
  float4 m;
  m.x = fmaxf(hv.x + a * w.x + b.x, 0.f);
  m.y = fmaxf(hv.y + a * w.y + b.y, 0.f);
  m.z = fmaxf(hv.z + a * w.z + b.z, 0.f);
  m.w = fmaxf(hv.w + a * w.w + b.w, 0.f);
  float* o = aggr3 + (long)dst * Hh + k0;
  atomicAdd(o + 0, m.x);
  atomicAdd(o + 1, m.y);
  atomicAdd(o + 2, m.z);
  atomicAdd(o + 3, m.w);
}

// ---------------------------------------------------------------------------
// WMMA GEMM: h3 = (hn + aggr3) @ W3^T + b3, fused with the decode projections:
//   p[n] = h3[n,:] . dw[0:128],  q[n] = h3[n,:] . dw[128:256]
// Block = 16 rows; 8 waves each own a 16-col tile; K = 128 in 4 steps of 32.
__global__ __launch_bounds__(256) void conv3_gemm_pq_kernel(
    const float* __restrict__ hn, const float* __restrict__ aggr3,
    const _Float16* __restrict__ W16, const float* __restrict__ b3,
    const float* __restrict__ dw,
    float* __restrict__ p, float* __restrict__ q)
{
  __shared__ _Float16 As[16][136];  // 16x128 strip, padded row stride (16B-aligned)
  int rowbase = blockIdx.x * 16;
  int tid = threadIdx.x;

  // Stage A: convert (hn + aggr3) -> f16 in LDS
  for (int i = tid; i < 16 * Hh; i += 256) {
    int r = i >> 7, k = i & (Hh - 1);
    long n = rowbase + r;
    As[r][k] = (_Float16)(hn[n * Hh + k] + aggr3[n * Hh + k]);
  }
  __syncthreads();

  int wave = tid >> 5;
  int lane = tid & 31;
  int g  = lane >> 4;   // lane half-group
  int hl = lane & 15;   // A: row m; B: col n
  int colbase = wave * 16;

  v8f acc = {};
#pragma unroll
  for (int kb = 0; kb < 4; ++kb) {
    int k0 = kb * 32;
    // A fragment: lane holds K = k0+16g..+15 (matches B order)
    v8h alo = *(const v8h*)(&As[hl][k0 + 16 * g]);
    v8h ahi = *(const v8h*)(&As[hl][k0 + 16 * g + 8]);
    v16h af;
#pragma unroll
    for (int i = 0; i < 8; ++i) { af[i] = alo[i]; af[i + 8] = ahi[i]; }
    // B fragment: column (colbase+hl), K = k0 + 16g .. +15 (32B contiguous)
    v16h bf = *(const v16h*)(W16 + (long)(colbase + hl) * Hh + k0 + 16 * g);
    acc = __builtin_amdgcn_wmma_f32_16x16x32_f16(
        false, af, false, bf, (short)0, acc, false, false);
  }

  // Epilogue: C/D layout -> (m = r + 8g, n = hl). Fold bias + decode dots.
  int col = colbase + hl;
  float d0 = dw[col], d1 = dw[Hh + col], bb = b3[col];
#pragma unroll
  for (int r = 0; r < 8; ++r) {
    int row = rowbase + r + 8 * g;
    float val = acc[r] + bb;
    float pv = val * d0;
    float qv = val * d1;
    // reduce across the 16 column lanes (xor<16 stays inside the half-group)
#pragma unroll
    for (int mask = 1; mask < 16; mask <<= 1) {
      pv += __shfl_xor(pv, mask, 32);
      qv += __shfl_xor(qv, mask, 32);
    }
    if (hl == 0) {
      atomicAdd(&p[row], pv);
      atomicAdd(&q[row], qv);
    }
  }
}

// ---------------------------------------------------------------------------
// Edge decode: out[e] = p[src] + q[dst] + db
__global__ __launch_bounds__(256) void decode_kernel(
    const int* __restrict__ ei, const float* __restrict__ p,
    const float* __restrict__ q, const float* __restrict__ db,
    float* __restrict__ out)
{
  int e = blockIdx.x * 256 + threadIdx.x;
  if (e >= Ee) return;
  out[e] = p[ei[e]] + q[ei[Ee + e]] + db[0];
}

// ---------------------------------------------------------------------------
extern "C" void kernel_launch(void* const* d_in, const int* in_sizes, int n_in,
                              void* d_out, int out_size, void* d_ws, size_t ws_size,
                              hipStream_t stream) {
  const float* x     = (const float*)d_in[0];
  const float* ea    = (const float*)d_in[1];
  const int*   ei    = (const int*)d_in[2];
  const float* emw   = (const float*)d_in[3];
  const float* emb   = (const float*)d_in[4];
  const float* lin_w = (const float*)d_in[5];
  const float* lin_b = (const float*)d_in[6];
  const float* nn1_w = (const float*)d_in[7];
  const float* nn1_b = (const float*)d_in[8];
  const float* w_ih  = (const float*)d_in[9];
  // d_in[10] = lstm_w_hh: dead (h0 == 0)
  const float* b_ih  = (const float*)d_in[11];
  const float* b_hh  = (const float*)d_in[12];
  const float* w3    = (const float*)d_in[13];
  const float* b3    = (const float*)d_in[14];
  const float* dw    = (const float*)d_in[15];
  const float* db    = (const float*)d_in[16];

  float* out_e  = (float*)d_out;             // [E]
  float* out_hn = out_e + Ee;                // [N*H]  (hn[None])
  float* out_c  = out_hn + (long)Nn * Hh;    // [N*H]  (c[None])

  float* ws    = (float*)d_ws;
  float* aggr1 = ws;                         // [N]
  float* p     = ws + Nn;                    // [N]
  float* q     = ws + 2 * Nn;                // [N]
  float* aggr3 = ws + 3 * Nn;                // [N*H]
  float* u     = aggr3 + (long)Nn * Hh;      // [512]
  float* v     = u + 4 * Hh;                 // [512]
  float* scal  = v + 4 * Hh;                 // [2] (+pad)
  _Float16* W16 = (_Float16*)(scal + 16);    // [128*128]

  long zcount = 3L * Nn + (long)Nn * Hh;     // aggr1, p, q, aggr3
  zero_ws_kernel<<<4096, 256, 0, stream>>>(ws, zcount);
  prep_kernel<<<1, 512, 0, stream>>>(emw, emb, lin_w, lin_b, nn1_w, nn1_b,
                                     w_ih, b_ih, b_hh, w3, u, v, scal, W16);
  edge1_kernel<<<(Ee + 255) / 256, 256, 0, stream>>>(x, ea, ei, scal, aggr1);
  node_lstm_kernel<<<(Nn * Hh) / 256, 256, 0, stream>>>(x, aggr1, u, v, out_hn, out_c);
  edge2_kernel<<<(Ee + 7) / 8, 256, 0, stream>>>(out_hn, ea, ei, emw, emb, aggr3);
  conv3_gemm_pq_kernel<<<Nn / 16, 256, 0, stream>>>(out_hn, aggr3, W16, b3, dw, p, q);
  decode_kernel<<<(Ee + 255) / 256, 256, 0, stream>>>(ei, p, q, db, out_e);
}
